// MiniGRU_40295383171508
// MI455X (gfx1250) — compile-verified, matrix operand order checked by reading, not run on hardware
//
#include <hip/hip_runtime.h>
#include <hip/hip_bf16.h>

// ---------------- problem constants (from reference setup_inputs) -----------
#define BS   4
#define SEQ  4096
#define DIN  1024
#define HH   1024
#define MTOT (BS * SEQ)          // 16384 rows
#define CHUNK 64                 // scan chunk length
#define NCH  (SEQ / CHUNK)       // 64 chunks per sequence

#define LDS_ROW 40               // padded LDS row stride in bf16 (80 B = 20 banks)

typedef __attribute__((ext_vector_type(16))) __bf16 v16bf;
typedef __attribute__((ext_vector_type(8)))  float  v8f;
typedef __attribute__((ext_vector_type(4)))  int    v4i;

// async global->LDS b128 builtin wants (int4 AS1*, int4 AS3*, imm, imm)
#define ASYNC_SRC(p) ((__attribute__((address_space(1))) v4i*)(v4i*)(p))
#define ASYNC_DST(p) ((__attribute__((address_space(3))) v4i*)(v4i*)(p))

// ---------------- math helpers ---------------------------------------------
__device__ __forceinline__ float softplusf(float v) {
    return fmaxf(v, 0.f) + log1pf(expf(-fabsf(v)));    // stable softplus
}
__device__ __forceinline__ float log_gf(float v) {
    // log g(v): v>=0 -> log(v+0.5), v<0 -> log sigmoid(v) = -softplus(-v)
    return (v >= 0.f) ? logf(v + 0.5f) : -softplusf(-v);
}
__device__ __forceinline__ float logaddexpf(float a, float b) {
    float m = fmaxf(a, b);
    return m + log1pf(expf(-fabsf(a - b)));
}

// ---------------- bf16 fragment loader (ISA 7.12.2 16-bit A/B layout) -------
// Per lane: row = (lane & 15), khalf = (lane >> 4).
// VGPRs 0..3 hold K = 8*khalf + 0..7 ; VGPRs 4..7 hold K = 16 + 8*khalf + 0..7
// => two 16-byte chunks at element offsets +0 and +16 from (row, 8*khalf).
union FragU { v16bf v; uint4 q[2]; };
__device__ __forceinline__ v16bf load_frag(const __bf16* p) {
    FragU f;
    f.q[0] = *(const uint4*)(p);
    f.q[1] = *(const uint4*)(p + 16);
    return f.v;
}

// ---------------- f32 -> bf16 packing kernels -------------------------------
__global__ __launch_bounds__(256) void cvt_x(const float* __restrict__ x,
                                             __bf16* __restrict__ xb) {
    size_t i = ((size_t)blockIdx.x * 256 + threadIdx.x) * 4;
    float4 f = *(const float4*)(x + i);
    union { __bf16 h[4]; uint2 u; } o;
    o.h[0] = (__bf16)f.x; o.h[1] = (__bf16)f.y;
    o.h[2] = (__bf16)f.z; o.h[3] = (__bf16)f.w;
    *(uint2*)(xb + i) = o.u;
}

__global__ __launch_bounds__(256) void cvt_w(const float* __restrict__ Wz,
                                             const float* __restrict__ Wh,
                                             __bf16* __restrict__ wc) {
    size_t i = ((size_t)blockIdx.x * 256 + threadIdx.x) * 4;    // over 2M elems
    const size_t half = (size_t)HH * DIN;
    const float* src = (i < half) ? (Wz + i) : (Wh + (i - half));
    float4 f = *(const float4*)src;
    union { __bf16 h[4]; uint2 u; } o;
    o.h[0] = (__bf16)f.x; o.h[1] = (__bf16)f.y;
    o.h[2] = (__bf16)f.z; o.h[3] = (__bf16)f.w;
    *(uint2*)(wc + i) = o.u;
}

// ---------------- fused dual-GEMM + log-space epilogue ----------------------
// Block tile: 64(M) x 128(N), 8 waves each owning a 16-col strip.
// The 64x32 bf16 A-tile is shared by all 8 waves -> stage it in LDS via
// async global->LDS DMA (ASYNCcnt), double-buffered across the K loop.
__global__ __launch_bounds__(256) void minigru_gemm(
    const __bf16* __restrict__ xb,   // [MTOT, DIN] bf16
    const __bf16* __restrict__ wc,   // [2*HH, DIN] bf16: rows 0..H-1 = Wz, H..2H-1 = Wh
    const float*  __restrict__ bz,
    const float*  __restrict__ bh,
    float* __restrict__ lcW,         // [MTOT, HH] log_coeff
    float* __restrict__ lvOut)       // [MTOT, HH] log_value (lives in d_out)
{
    __shared__ __bf16 atile[2][64 * LDS_ROW];   // 2 x 5120 B double buffer

    const int lane  = threadIdx.x & 31;
    const int wave  = threadIdx.x >> 5;
    const int mbase = blockIdx.x * 64;
    const int nbase = (blockIdx.y * 8 + wave) * 16;
    const int rl    = lane & 15;   // row-within-tile (A: M, B: N, C/D: N)
    const int kg    = lane >> 4;   // K-half selector

    // async A-tile copy: thread t moves one 16-byte chunk per K-step
    const int crow   = threadIdx.x >> 2;       // 0..63
    const int cchunk = threadIdx.x & 3;        // 0..3 (4 x 8 bf16 per row)
    const __bf16* gsrc = xb + (size_t)(mbase + crow) * DIN + cchunk * 8;
    __bf16* ldst0 = &atile[0][crow * LDS_ROW + cchunk * 8];
    __bf16* ldst1 = &atile[1][crow * LDS_ROW + cchunk * 8];

    // B fragments: per-wave weight rows (L2-resident), global path
    const __bf16* bpz = wc + (size_t)(nbase + rl) * DIN + 8 * kg;
    const __bf16* bph = wc + (size_t)(HH + nbase + rl) * DIN + 8 * kg;

    // LDS A-fragment base for this lane (rows 16*i + rl, K-half kg)
    const __bf16* lfrag0 = &atile[0][rl * LDS_ROW + 8 * kg];
    const __bf16* lfrag1 = &atile[1][rl * LDS_ROW + 8 * kg];

    v8f accz[4] = {}, acch[4] = {};

    // preload tile 0
    __builtin_amdgcn_global_load_async_to_lds_b128(ASYNC_SRC(gsrc), ASYNC_DST(ldst0), 0, 0);

    #pragma unroll
    for (int i = 0; i < DIN / 32; ++i) {
        const int kk = i * 32;
        const __bf16* lf = (i & 1) ? lfrag1 : lfrag0;
        if (i + 1 < DIN / 32) {
            __builtin_amdgcn_global_load_async_to_lds_b128(
                ASYNC_SRC(gsrc + kk + 32),
                ASYNC_DST((i & 1) ? ldst0 : ldst1), 0, 0);
            __builtin_prefetch(bpz + kk + 128);          // weights: L2 warm-up
            asm volatile("s_wait_asynccnt 0x1" ::: "memory");   // tile i landed
        } else {
            asm volatile("s_wait_asynccnt 0x0" ::: "memory");
        }
        __syncthreads();                                  // whole block's DMA done

        v16bf bzf = load_frag(bpz + kk);
        v16bf bhf = load_frag(bph + kk);
        v16bf a0 = load_frag(lf + 0 * 16 * LDS_ROW);      // ds_load_b128 x2 each
        v16bf a1 = load_frag(lf + 1 * 16 * LDS_ROW);
        v16bf a2 = load_frag(lf + 2 * 16 * LDS_ROW);
        v16bf a3 = load_frag(lf + 3 * 16 * LDS_ROW);

        accz[0] = __builtin_amdgcn_wmma_f32_16x16x32_bf16(false, a0, false, bzf, (short)0, accz[0], false, false);
        acch[0] = __builtin_amdgcn_wmma_f32_16x16x32_bf16(false, a0, false, bhf, (short)0, acch[0], false, false);
        accz[1] = __builtin_amdgcn_wmma_f32_16x16x32_bf16(false, a1, false, bzf, (short)0, accz[1], false, false);
        acch[1] = __builtin_amdgcn_wmma_f32_16x16x32_bf16(false, a1, false, bhf, (short)0, acch[1], false, false);
        accz[2] = __builtin_amdgcn_wmma_f32_16x16x32_bf16(false, a2, false, bzf, (short)0, accz[2], false, false);
        acch[2] = __builtin_amdgcn_wmma_f32_16x16x32_bf16(false, a2, false, bhf, (short)0, acch[2], false, false);
        accz[3] = __builtin_amdgcn_wmma_f32_16x16x32_bf16(false, a3, false, bzf, (short)0, accz[3], false, false);
        acch[3] = __builtin_amdgcn_wmma_f32_16x16x32_bf16(false, a3, false, bhf, (short)0, acch[3], false, false);

        __syncthreads();   // all waves done reading buffer before it is reused
    }

    // C/D layout: lanes 0-15 -> N=lane, M=vgpr j ; lanes 16-31 -> N=lane-16, M=j+8
    const int h = nbase + rl;
    const float bzv = bz[h], bhv = bh[h];
    #pragma unroll
    for (int i = 0; i < 4; ++i) {
        #pragma unroll
        for (int j = 0; j < 8; ++j) {
            const int m = mbase + 16 * i + j + 8 * kg;
            const float kv = accz[i][j] + bzv;            // gate pre-activation
            const float pv = acch[i][j] + bhv;            // candidate pre-activation
            const float sp = softplusf(kv);
            const float lc = -sp;                         // log(1-z)
            const float lv = (kv - sp) + log_gf(pv);      // log(z) + log g(p)
            const size_t idx = (size_t)m * HH + h;
            lcW[idx]   = lc;
            lvOut[idx] = lv;
        }
    }
}

// ---------------- chunked log-space linear-recurrence scan ------------------
// Pass 1: per-chunk summary (A = sum lc, B = streaming logaddexp composition).
__global__ __launch_bounds__(256) void scan_chunks(const float* __restrict__ lcW,
                                                   const float* __restrict__ lvOut,
                                                   float* __restrict__ cA,
                                                   float* __restrict__ cB) {
    const int idx = blockIdx.x * 256 + threadIdx.x;   // ((b*NCH + c)*HH + h)
    const int h = idx & (HH - 1);
    const int c = (idx >> 10) & (NCH - 1);
    const int b = idx >> 16;
    size_t g = ((size_t)(b * SEQ + c * CHUNK)) * HH + h;
    float A = 0.f, B = -INFINITY;
    for (int t = 0; t < CHUNK; ++t, g += HH) {
        const float lc = lcW[g], lv = lvOut[g];
        A += lc;
        B = logaddexpf(lv, lc + B);
    }
    cA[idx] = A; cB[idx] = B;
}

// Pass 2: serial scan over the 64 chunk summaries per sequence (4096 threads).
__global__ __launch_bounds__(256) void scan_carry(const float* __restrict__ cA,
                                                  const float* __restrict__ cB,
                                                  const float* __restrict__ h0,
                                                  float* __restrict__ carry) {
    const int idx = blockIdx.x * 256 + threadIdx.x;   // b*HH + h
    const int h = idx & (HH - 1);
    const int b = idx >> 10;
    float cr = log_gf(h0[idx]);                       // h_0 is [BS,1,HH]
    for (int ci = 0; ci < NCH; ++ci) {
        const int k = (b * NCH + ci) * HH + h;
        carry[k] = cr;
        cr = logaddexpf(cB[k], cA[k] + cr);
    }
}

// Pass 3: replay each chunk from its carry; overwrite d_out with exp(log h_t).
__global__ __launch_bounds__(256) void scan_apply(const float* __restrict__ lcW,
                                                  const float* __restrict__ carry,
                                                  float* __restrict__ lvOut) {
    const int idx = blockIdx.x * 256 + threadIdx.x;
    const int h = idx & (HH - 1);
    const int c = (idx >> 10) & (NCH - 1);
    const int b = idx >> 16;
    float lh = carry[idx];
    size_t g = ((size_t)(b * SEQ + c * CHUNK)) * HH + h;
    for (int t = 0; t < CHUNK; ++t, g += HH) {
        const float lc = lcW[g], lv = lvOut[g];
        lh = logaddexpf(lv, lc + lh);
        lvOut[g] = expf(lh);
    }
}

// ---------------- launch ----------------------------------------------------
extern "C" void kernel_launch(void* const* d_in, const int* in_sizes, int n_in,
                              void* d_out, int out_size, void* d_ws, size_t ws_size,
                              hipStream_t stream) {
    const float* x   = (const float*)d_in[0];
    const float* h0  = (const float*)d_in[1];
    const float* Wz  = (const float*)d_in[2];
    const float* bz  = (const float*)d_in[3];
    const float* Wh  = (const float*)d_in[4];
    const float* bh  = (const float*)d_in[5];
    float* out = (float*)d_out;

    char* ws = (char*)d_ws;
    __bf16* Wc  = (__bf16*)(ws);                              //  4 MiB: [2H, D] bf16
    __bf16* Xb  = (__bf16*)(ws + ((size_t)4   << 20));        // 32 MiB: [M, D]  bf16
    float*  lcW = (float*) (ws + ((size_t)36  << 20));        // 64 MiB: [M, H]  f32
    float*  cA  = (float*) (ws + ((size_t)100 << 20));        //  1 MiB
    float*  cB  = (float*) (ws + ((size_t)101 << 20));        //  1 MiB
    float*  cry = (float*) (ws + ((size_t)102 << 20));        //  1 MiB

    // 1) pack inputs to bf16
    cvt_x<<<(MTOT * DIN) / (4 * 256), 256, 0, stream>>>(x, Xb);
    cvt_w<<<(2 * HH * DIN) / (4 * 256), 256, 0, stream>>>(Wz, Wh, Wc);

    // 2) fused dual-GEMM (WMMA bf16, async LDS staging) + log-space epilogue
    minigru_gemm<<<dim3(MTOT / 64, HH / 128), 256, 0, stream>>>(Xb, Wc, bz, bh, lcW, out);

    // 3) chunked parallel scan
    scan_chunks<<<(BS * NCH * HH) / 256, 256, 0, stream>>>(lcW, out, cA, cB);
    scan_carry<<<(BS * HH) / 256, 256, 0, stream>>>(cA, cB, h0, cry);
    scan_apply<<<(BS * NCH * HH) / 256, 256, 0, stream>>>(lcW, cry, out);
}
